// WindowAttention3D_52639119180252
// MI455X (gfx1250) — compile-verified, matrix operand order checked by reading, not run on hardware
//
#include <hip/hip_runtime.h>

// CDNA5 / gfx1250 window-attention pipeline, bf16 WMMA with f32 accumulation.
// v3: two N-tiles per GEMM iteration (dual accumulator chains -> more loads in
// flight between s_wait_loadcnt), division-free epilogue index mapping,
// LDS-staged k/v panels in attention, global_prefetch of upcoming weight rows.

typedef __attribute__((ext_vector_type(16))) __bf16 v16bf;
typedef __attribute__((ext_vector_type(8)))  float  v8f;

#define WIN_N  64      // tokens per 4x4x4 window
#define CDIM   384
#define NHEADS 8
#define HD     48      // head dim
#define HDP    64      // head dim padded to K-multiple of 32
#define NWIN   1024    // 2 * 8*8*8 windows

union FragBF {
  v16bf v;
  uint4 q[2];
};

// Load one 16x32 bf16 WMMA A/B fragment. Lane l holds row (l&15); its 16
// K-values are two contiguous 8-element chunks at koff and koff+16, where
// koff already includes the 8*(lane>>4) half-wave shift (ISA 16-bit A layout).
__device__ __forceinline__ v16bf load_frag(const __bf16* rowp, int koff) {
  FragBF f;
  f.q[0] = *reinterpret_cast<const uint4*>(rowp + koff);
  f.q[1] = *reinterpret_cast<const uint4*>(rowp + koff + 16);
  return f.v;
}

__device__ __forceinline__ v8f wmma_bf16(v16bf a, v16bf b, v8f c) {
  return __builtin_amdgcn_wmma_f32_16x16x32_bf16(
      /*neg_a=*/false, a, /*neg_b=*/false, b,
      /*c_mod=*/(short)0, c, /*reuse_a=*/false, /*reuse_b=*/false);
}

// ---------------- stage 0: window partition x -> xw (win, tok, C) bf16 ------
__global__ void k_winpart(const float* __restrict__ x, __bf16* __restrict__ xw,
                          int total) {
  int i = blockIdx.x * blockDim.x + threadIdx.x;
  if (i >= total) return;
  int w = i & 31, h = (i >> 5) & 31, d = (i >> 10) & 31;
  int bc = i >> 15;               // b*384 + c
  int c = bc % CDIM;
  int b = bc / CDIM;
  int win = ((b * 8 + (d >> 2)) * 8 + (h >> 2)) * 8 + (w >> 2);
  int tok = ((d & 3) << 4) + ((h & 3) << 2) + (w & 3);
  xw[(win * WIN_N + tok) * CDIM + c] = (__bf16)x[i];
}

__global__ void k_cvt(const float* __restrict__ s, __bf16* __restrict__ d, int n) {
  int i = blockIdx.x * blockDim.x + threadIdx.x;
  if (i < n) d[i] = (__bf16)s[i];
}

__global__ void k_zero4(uint4* __restrict__ p, int n) {
  int i = blockIdx.x * blockDim.x + threadIdx.x;
  if (i < n) p[i] = make_uint4(0u, 0u, 0u, 0u);
}

// ---------------- stage 1: QKV GEMM: (64x384) @ qkv_w^T (384x1152) ----------
__global__ __launch_bounds__(128) void k_qkv(
    const __bf16* __restrict__ xw, const __bf16* __restrict__ wq,
    const float* __restrict__ bias, __bf16* __restrict__ qb,
    __bf16* __restrict__ kb, __bf16* __restrict__ vtb) {
  const int win  = blockIdx.x;
  const int lane = threadIdx.x & 31;
  const int wave = threadIdx.x >> 5;        // M-tile (16 tokens) per wave
  const int colL = lane & 15;
  const int koffL = (lane >> 4) << 3;       // A/B half-wave K shift, also row half

  // Preload full 16x384 A row-block (12 K-tiles) in registers.
  const __bf16* arow = xw + (win * WIN_N + wave * 16 + colL) * CDIM;
  v16bf a[12];
#pragma unroll
  for (int kt = 0; kt < 12; ++kt) a[kt] = load_frag(arow, kt * 32 + koffL);

  // 72 N-tiles = 3 parts (q,k,v) x 8 heads x 3 j-tiles; two tiles/iteration.
#pragma unroll 1
  for (int part = 0; part < 3; ++part) {
    __bf16* qk = (part == 0) ? qb : kb;     // used when part < 2
    int hh = 0, jt = 0;                     // incremental (head, j-tile) counter
#pragma unroll 1
    for (int p = 0; p < 12; ++p) {
      const int base = (part * 24 + 2 * p) * 16;   // first column of tile pair
      const __bf16* b0row = wq + (base + colL) * CDIM;
      const __bf16* b1row = b0row + 16 * CDIM;
      if (p < 11) __builtin_prefetch(b0row + 32 * CDIM, 0, 3);  // next pair

      v8f acc0 = {}, acc1 = {};
#pragma unroll
      for (int kt = 0; kt < 12; ++kt) {
        acc0 = wmma_bf16(a[kt], load_frag(b0row, kt * 32 + koffL), acc0);
        acc1 = wmma_bf16(a[kt], load_frag(b1row, kt * 32 + koffL), acc1);
      }

      const float bv0 = bias[base + colL];
      const float bv1 = bias[base + 16 + colL];

      // epilogue tile 0: head hh, j = jt*16 + colL
      {
        const int j = jt * 16 + colL;
#pragma unroll
        for (int r = 0; r < 8; ++r) {
          const int tok = wave * 16 + r + koffL;
          const float val = acc0[r] + bv0;
          if (part < 2)
            qk[((win * NHEADS + hh) * WIN_N + tok) * HDP + j] = (__bf16)val;
          else
            vtb[((win * NHEADS + hh) * HDP + j) * WIN_N + tok] = (__bf16)val;
        }
        if (++jt == 3) { jt = 0; ++hh; }
      }
      // epilogue tile 1
      {
        const int j = jt * 16 + colL;
#pragma unroll
        for (int r = 0; r < 8; ++r) {
          const int tok = wave * 16 + r + koffL;
          const float val = acc1[r] + bv1;
          if (part < 2)
            qk[((win * NHEADS + hh) * WIN_N + tok) * HDP + j] = (__bf16)val;
          else
            vtb[((win * NHEADS + hh) * HDP + j) * WIN_N + tok] = (__bf16)val;
        }
        if (++jt == 3) { jt = 0; ++hh; }
      }
    }
  }
}

// ---------------- stage 2: attention per (window, head) ---------------------
__global__ __launch_bounds__(128) void k_attn(
    const __bf16* __restrict__ qb, const __bf16* __restrict__ kb,
    const __bf16* __restrict__ vtb, __bf16* __restrict__ obuf) {
  __shared__ __align__(16) __bf16 kls[WIN_N * HDP];     // 8 KB: k panel
  __shared__ __align__(16) __bf16 vls[HD * WIN_N];      // 6 KB: v^T panel
  __shared__ __align__(16) __bf16 pls[4][16 * WIN_N];   // 8 KB: P panels
  const int wh  = blockIdx.x;
  const int win = wh >> 3, hh = wh & 7;
  const int lane = threadIdx.x & 31, wave = threadIdx.x >> 5;
  const int colL = lane & 15, koffL = (lane >> 4) << 3;
  const float scale = 0.14433756729740643f;  // 48^-0.5

  const __bf16* qh = qb  + (size_t)(win * NHEADS + hh) * WIN_N * HDP;
  const __bf16* kh = kb  + (size_t)(win * NHEADS + hh) * WIN_N * HDP;
  const __bf16* vh = vtb + (size_t)(win * NHEADS + hh) * HDP * WIN_N;

  // Stage k and v^T panels in LDS once per block (they are reused by all
  // 4 waves x several tiles). Coalesced 16B copies by all 128 threads.
  {
    const uint4* ksrc = reinterpret_cast<const uint4*>(kh);
    uint4* kdst = reinterpret_cast<uint4*>(kls);
#pragma unroll
    for (int i = 0; i < 4; ++i)                 // 512 uint4
      kdst[threadIdx.x + 128 * i] = ksrc[threadIdx.x + 128 * i];
    const uint4* vsrc = reinterpret_cast<const uint4*>(vh);
    uint4* vdst = reinterpret_cast<uint4*>(vls);
#pragma unroll
    for (int i = 0; i < 3; ++i)                 // 384 uint4
      vdst[threadIdx.x + 128 * i] = vsrc[threadIdx.x + 128 * i];
  }

  // S = q @ k^T  (K = 64, hd padded with zeros); q rows loaded from global
  // (each row used by exactly one wave).
  const __bf16* arow = qh + (wave * 16 + colL) * HDP;
  v16bf aq0 = load_frag(arow, koffL);
  v16bf aq1 = load_frag(arow, 32 + koffL);

  __syncthreads();   // k/v panels visible

  v8f s[4];
#pragma unroll
  for (int nt = 0; nt < 4; ++nt) {
    const __bf16* brow = kls + (nt * 16 + colL) * HDP;
    v8f acc = {};
    acc = wmma_bf16(aq0, load_frag(brow, koffL), acc);
    acc = wmma_bf16(aq1, load_frag(brow, 32 + koffL), acc);
    s[nt] = acc * scale;
  }

  // Row softmax in f32. Row r (resp. r+8) lives in VGPR r of lanes 0-15
  // (resp. 16-31); xor-shuffles with mask<16 reduce the two halves independently.
#pragma unroll
  for (int r = 0; r < 8; ++r) {
    float m = s[0][r];
#pragma unroll
    for (int nt = 1; nt < 4; ++nt) m = fmaxf(m, s[nt][r]);
#pragma unroll
    for (int off = 1; off < 16; off <<= 1) m = fmaxf(m, __shfl_xor(m, off, 32));
    float sum = 0.f;
#pragma unroll
    for (int nt = 0; nt < 4; ++nt) {
      float e = __expf(s[nt][r] - m);
      s[nt][r] = e;
      sum += e;
    }
#pragma unroll
    for (int off = 1; off < 16; off <<= 1) sum += __shfl_xor(sum, off, 32);
    const float inv = 1.0f / sum;
#pragma unroll
    for (int nt = 0; nt < 4; ++nt) s[nt][r] *= inv;
  }

  // C-layout -> A-layout via LDS (per-wave private 16x64 bf16 panel).
#pragma unroll
  for (int nt = 0; nt < 4; ++nt)
#pragma unroll
    for (int r = 0; r < 8; ++r)
      pls[wave][(r + koffL) * WIN_N + nt * 16 + colL] = (__bf16)s[nt][r];
  __syncthreads();

  const __bf16* prow = &pls[wave][colL * WIN_N];
  v16bf pa0 = load_frag(prow, koffL);
  v16bf pa1 = load_frag(prow, 32 + koffL);

  // O = P @ v : N = hd = 48 -> 3 N-tiles; B column j == vT row j (in LDS).
#pragma unroll
  for (int nt = 0; nt < 3; ++nt) {
    const __bf16* brow = vls + (nt * 16 + colL) * WIN_N;
    v8f acc = {};
    acc = wmma_bf16(pa0, load_frag(brow, koffL), acc);
    acc = wmma_bf16(pa1, load_frag(brow, 32 + koffL), acc);
    const int ch = hh * HD + nt * 16 + colL;
#pragma unroll
    for (int r = 0; r < 8; ++r) {
      const int tok = wave * 16 + r + koffL;
      obuf[(win * WIN_N + tok) * CDIM + ch] = (__bf16)acc[r];
    }
  }
}

// ---------------- stage 3: proj GEMM + un-window scatter ---------------------
__global__ __launch_bounds__(128) void k_proj(
    const __bf16* __restrict__ obuf, const __bf16* __restrict__ pw,
    const float* __restrict__ pb, float* __restrict__ out) {
  const int win = blockIdx.x;
  const int lane = threadIdx.x & 31, wave = threadIdx.x >> 5;
  const int colL = lane & 15, koffL = (lane >> 4) << 3;

  const __bf16* arow = obuf + (win * WIN_N + wave * 16 + colL) * CDIM;
  v16bf a[12];
#pragma unroll
  for (int kt = 0; kt < 12; ++kt) a[kt] = load_frag(arow, kt * 32 + koffL);

  const int b  = win >> 9;
  const int dq = (win >> 6) & 7, hq = (win >> 3) & 7, wq = win & 7;

  // 24 N-tiles, two per iteration with dual accumulator chains.
#pragma unroll 1
  for (int p = 0; p < 12; ++p) {
    const int base = 2 * p * 16;
    const __bf16* b0row = pw + (base + colL) * CDIM;
    const __bf16* b1row = b0row + 16 * CDIM;
    if (p < 11) __builtin_prefetch(b0row + 32 * CDIM, 0, 3);

    v8f acc0 = {}, acc1 = {};
#pragma unroll
    for (int kt = 0; kt < 12; ++kt) {
      acc0 = wmma_bf16(a[kt], load_frag(b0row, kt * 32 + koffL), acc0);
      acc1 = wmma_bf16(a[kt], load_frag(b1row, kt * 32 + koffL), acc1);
    }

    const float bv0 = pb[base + colL];
    const float bv1 = pb[base + 16 + colL];
#pragma unroll
    for (int r = 0; r < 8; ++r) {
      const int tok = wave * 16 + r + koffL;
      const int dl = tok >> 4, hl = (tok >> 2) & 3, wl = tok & 3;
      const int d = dq * 4 + dl, hc = hq * 4 + hl, wc = wq * 4 + wl;
      const size_t sp = (size_t)((d << 10) + (hc << 5) + wc);
      out[((size_t)(b * CDIM + base + colL) << 15) + sp] = acc0[r] + bv0;
      out[((size_t)(b * CDIM + base + 16 + colL) << 15) + sp] = acc1[r] + bv1;
    }
  }
}

// ---------------------------------------------------------------------------
extern "C" void kernel_launch(void* const* d_in, const int* in_sizes, int n_in,
                              void* d_out, int out_size, void* d_ws,
                              size_t ws_size, hipStream_t stream) {
  const float* x      = (const float*)d_in[0];
  const float* qkv_w  = (const float*)d_in[1];
  const float* qkv_b  = (const float*)d_in[2];
  const float* proj_w = (const float*)d_in[3];
  const float* proj_b = (const float*)d_in[4];
  float* out = (float*)d_out;

  char* ws = (char*)d_ws;
  size_t off = 0;
  __bf16* xw  = (__bf16*)(ws + off); off += (size_t)NWIN * WIN_N * CDIM * 2;   // 48 MB
  __bf16* wqb = (__bf16*)(ws + off); off += (size_t)3 * CDIM * CDIM * 2;       // qkv_w bf16
  __bf16* pwb = (__bf16*)(ws + off); off += (size_t)CDIM * CDIM * 2;           // proj_w bf16
  __bf16* qb  = (__bf16*)(ws + off); off += (size_t)NWIN * NHEADS * WIN_N * HDP * 2; // 64 MB
  __bf16* kb  = (__bf16*)(ws + off); off += (size_t)NWIN * NHEADS * WIN_N * HDP * 2; // 64 MB
  __bf16* vtb = (__bf16*)(ws + off); off += (size_t)NWIN * NHEADS * HDP * WIN_N * 2; // 64 MB
  __bf16* obuf = xw;  // xw is dead after k_qkv; reuse for attention output

  const int totalX = 2 * CDIM * 32 * 32 * 32;  // 25,165,824
  k_winpart<<<dim3((totalX + 255) / 256), dim3(256), 0, stream>>>(x, xw, totalX);
  k_cvt<<<dim3((3 * CDIM * CDIM + 255) / 256), dim3(256), 0, stream>>>(
      qkv_w, wqb, 3 * CDIM * CDIM);
  k_cvt<<<dim3((CDIM * CDIM + 255) / 256), dim3(256), 0, stream>>>(
      proj_w, pwb, CDIM * CDIM);
  // zero q and k buffers (contiguous) so the hd 48->64 pad is exact zero
  const int nzero = (int)(((size_t)2 * NWIN * NHEADS * WIN_N * HDP * 2) / 16);
  k_zero4<<<dim3((nzero + 255) / 256), dim3(256), 0, stream>>>((uint4*)qb, nzero);

  k_qkv<<<dim3(NWIN), dim3(128), 0, stream>>>(xw, wqb, qkv_b, qb, kb, vtb);
  k_attn<<<dim3(NWIN * NHEADS), dim3(128), 0, stream>>>(qb, kb, vtb, obuf);
  k_proj<<<dim3(NWIN), dim3(128), 0, stream>>>(obuf, pwb, proj_b, out);
}
